// SelfAttention_26706106646794
// MI455X (gfx1250) — compile-verified, matrix operand order checked by reading, not run on hardware
//
#include <hip/hip_runtime.h>
#include <hip/hip_bf16.h>

typedef __bf16 bf16_t;
typedef __attribute__((ext_vector_type(16))) __bf16 v16bf;
typedef __attribute__((ext_vector_type(8)))  __bf16 v8bf;
typedef __attribute__((ext_vector_type(8)))  float  v8f;

#define DD 1024
#define BB 128
#define EPSF 1e-5f
#define ISPLIT 4   // blocks per batch row in the attention kernel

// ---------------------------------------------------------------------------
// Kernel 1: fused scalar-affine Q/K/V + rank-1 softmax attention.
// Grid (BB, ISPLIT): each block rebuilds K,V for its batch row in LDS (8 KB),
// reduces row max/min once, then each of the 256 threads owns exactly ONE
// output position i (1024 v_exp_f32 per thread). 512 workgroups / 4096 waves
// keeps all WGPs of the MI455X busy on the transcendental-bound hot loop.
// Writes attended[b,i] as bf16 (A-matrix of the downstream WMMA GEMM).
// ---------------------------------------------------------------------------
__global__ void __launch_bounds__(256)
attn_kernel(const float* __restrict__ x,
            const float* __restrict__ wq, const float* __restrict__ bq,
            const float* __restrict__ wk, const float* __restrict__ bk,
            const float* __restrict__ wv, const float* __restrict__ bv,
            bf16_t* __restrict__ attended_bf)
{
    __shared__ float Ks[DD];
    __shared__ float Vs[DD];
    __shared__ float redmax[256];
    __shared__ float redmin[256];

    const int b   = blockIdx.x;
    const int tid = threadIdx.x;

    const float fwq = *wq, fbq = *bq;
    const float fwk = *wk, fbk = *bk;
    const float fwv = *wv, fbv = *bv;

    float lmax = -3.0e38f, lmin = 3.0e38f;
    for (int d = tid; d < DD; d += 256) {
        float xv = x[b * DD + d];
        float k  = xv * fwk + fbk;
        float v  = xv * fwv + fbv;
        Ks[d] = k;
        Vs[d] = v;
        lmax = fmaxf(lmax, k);
        lmin = fminf(lmin, k);
    }
    redmax[tid] = lmax;
    redmin[tid] = lmin;
    __syncthreads();
    for (int s = 128; s > 0; s >>= 1) {
        if (tid < s) {
            redmax[tid] = fmaxf(redmax[tid], redmax[tid + s]);
            redmin[tid] = fminf(redmin[tid], redmin[tid + s]);
        }
        __syncthreads();
    }
    const float kmax = redmax[0];
    const float kmin = redmin[0];

    // One output position per thread.
    const int i  = blockIdx.y * 256 + tid;
    const float qi = x[b * DD + i] * fwq + fbq;
    const float m  = (qi >= 0.f) ? qi * kmax : qi * kmin;  // exact row max
    float num = 0.f, den = 0.f;
    #pragma unroll 2
    for (int j = 0; j < DD; j += 4) {
        float4 kk = *(const float4*)(Ks + j);
        float4 vv = *(const float4*)(Vs + j);
        float e0 = __expf(qi * kk.x - m);
        float e1 = __expf(qi * kk.y - m);
        float e2 = __expf(qi * kk.z - m);
        float e3 = __expf(qi * kk.w - m);
        num += e0 * vv.x + e1 * vv.y + e2 * vv.z + e3 * vv.w;
        den += (e0 + e1) + (e2 + e3);
    }
    attended_bf[b * DD + i] = (bf16_t)(num / den);
}

// ---------------------------------------------------------------------------
// Kernel 2: Wout f32 -> bf16 (one-time layout prep for the WMMA B-matrix).
// ---------------------------------------------------------------------------
__global__ void __launch_bounds__(256)
cvt_bf16_kernel(const float* __restrict__ src, bf16_t* __restrict__ dst)
{
    const int i = (blockIdx.x * 256 + threadIdx.x) * 4;
    float4 f = *(const float4*)(src + i);
    dst[i + 0] = (bf16_t)f.x;
    dst[i + 1] = (bf16_t)f.y;
    dst[i + 2] = (bf16_t)f.z;
    dst[i + 3] = (bf16_t)f.w;
}

// ---------------------------------------------------------------------------
// Kernel 3: proj = attended @ Wout^T + bout via v_wmma_f32_16x16x32_bf16.
// One wave per 16x16 output tile, K-loop of 32 steps. Fragment layout per
// CDNA5 ISA 7.12.2: A/B 16-bit — lane%16 = M/N, lane/16 = h; elems 0..7 hold
// K = 8h..8h+7, elems 8..15 hold K = 16+8h..16+8h+7 (two b128 loads each).
// C/D: N = lane%16, M = 8*(lane/16) + vgpr.
// ---------------------------------------------------------------------------
__global__ void __launch_bounds__(256)
gemm_wmma_kernel(const bf16_t* __restrict__ A,     // [128 x 1024] bf16
                 const bf16_t* __restrict__ Wb,    // [1024 x 1024] bf16 (row o, fast k)
                 const float*  __restrict__ bout,  // [1024]
                 float*        __restrict__ proj)  // [128 x 1024] f32
{
    const int wave = threadIdx.x >> 5;
    const int lane = threadIdx.x & 31;
    const int tile = blockIdx.x * 8 + wave;   // 512 tiles total
    const int mt = tile & 7;                  // 8 tiles along M (=B rows)
    const int nt = tile >> 3;                 // 64 tiles along N (=out features)
    const int m0 = mt * 16;
    const int n0 = nt * 16;

    const int h = lane >> 4;                  // half-wave
    const int r = lane & 15;                  // M index for A, N index for B

    const bf16_t* pa = A  + (m0 + r) * DD;
    const bf16_t* pb = Wb + (n0 + r) * DD;    // proj contracts over Wout's fast axis

    v8f c = {};
    #pragma unroll 4
    for (int k0 = 0; k0 < DD; k0 += 32) {
        const bf16_t* pah = pa + k0 + 8 * h;
        const bf16_t* pbh = pb + k0 + 8 * h;
        v8bf a_lo = *(const v8bf*)(pah);
        v8bf a_hi = *(const v8bf*)(pah + 16);
        v8bf b_lo = *(const v8bf*)(pbh);
        v8bf b_hi = *(const v8bf*)(pbh + 16);
        v16bf av = __builtin_shufflevector(a_lo, a_hi,
                     0,1,2,3,4,5,6,7,8,9,10,11,12,13,14,15);
        v16bf bv = __builtin_shufflevector(b_lo, b_hi,
                     0,1,2,3,4,5,6,7,8,9,10,11,12,13,14,15);
        c = __builtin_amdgcn_wmma_f32_16x16x32_bf16(
                /*neg_a=*/false, av, /*neg_b=*/false, bv,
                /*c_mod=*/(short)0, c, /*reuse_a=*/false, /*reuse_b=*/false);
    }

    const float bias = bout[n0 + r];
    #pragma unroll
    for (int q = 0; q < 8; q++) {
        proj[(m0 + 8 * h + q) * DD + n0 + r] = c[q] + bias;
    }
}

// ---------------------------------------------------------------------------
// Kernel 4: residual + LayerNorm. One block per batch row.
// ---------------------------------------------------------------------------
__global__ void __launch_bounds__(256)
ln_kernel(const float* __restrict__ x, const float* __restrict__ proj,
          const float* __restrict__ gamma, const float* __restrict__ beta,
          float* __restrict__ out)
{
    __shared__ float red1[256];
    __shared__ float red2[256];
    const int b   = blockIdx.x;
    const int tid = threadIdx.x;

    float hv[4];
    float s1 = 0.f, s2 = 0.f;
    #pragma unroll
    for (int t = 0; t < 4; t++) {
        int d  = tid + t * 256;
        float h = x[b * DD + d] + proj[b * DD + d];
        hv[t] = h;
        s1 += h;
        s2 += h * h;
    }
    red1[tid] = s1;
    red2[tid] = s2;
    __syncthreads();
    for (int s = 128; s > 0; s >>= 1) {
        if (tid < s) {
            red1[tid] += red1[tid + s];
            red2[tid] += red2[tid + s];
        }
        __syncthreads();
    }
    const float mu   = red1[0] * (1.0f / DD);
    const float var  = red2[0] * (1.0f / DD) - mu * mu;
    const float rstd = rsqrtf(var + EPSF);
    #pragma unroll
    for (int t = 0; t < 4; t++) {
        int d = tid + t * 256;
        out[b * DD + d] = (hv[t] - mu) * rstd * gamma[d] + beta[d];
    }
}

// ---------------------------------------------------------------------------
extern "C" void kernel_launch(void* const* d_in, const int* in_sizes, int n_in,
                              void* d_out, int out_size, void* d_ws, size_t ws_size,
                              hipStream_t stream)
{
    const float* x     = (const float*)d_in[0];
    const float* wq    = (const float*)d_in[1];
    const float* bq    = (const float*)d_in[2];
    const float* wk    = (const float*)d_in[3];
    const float* bk    = (const float*)d_in[4];
    const float* wv    = (const float*)d_in[5];
    const float* bv    = (const float*)d_in[6];
    const float* Wout  = (const float*)d_in[7];
    const float* bout  = (const float*)d_in[8];
    const float* gamma = (const float*)d_in[9];
    const float* beta  = (const float*)d_in[10];
    float* out = (float*)d_out;

    // Workspace layout (all 256B-aligned offsets):
    //   [0,          256KB)  attended bf16  [128 x 1024]
    //   [256KB,     2.25MB)  Wout bf16      [1024 x 1024]
    //   [2.25MB,    2.75MB)  proj f32       [128 x 1024]
    char* ws = (char*)d_ws;
    bf16_t* attended_bf = (bf16_t*)(ws);
    bf16_t* wout_bf     = (bf16_t*)(ws + 256 * 1024);
    float*  proj        = (float*)(ws + 256 * 1024 + 2 * 1024 * 1024);

    // 1) fused attention (transcendental-bound: 134M v_exp_f32),
    //    512 workgroups to fill the WGP array
    dim3 agrid(BB, ISPLIT);
    attn_kernel<<<agrid, 256, 0, stream>>>(x, wq, bq, wk, bk, wv, bv, attended_bf);

    // 2) Wout -> bf16 (1M elems, 4 per thread)
    cvt_bf16_kernel<<<(DD * DD) / (256 * 4), 256, 0, stream>>>(Wout, wout_bf);

    // 3) WMMA GEMM + bias: 512 16x16 tiles, 8 waves/block -> 64 blocks
    gemm_wmma_kernel<<<64, 256, 0, stream>>>(attended_bf, wout_bf, bout, proj);

    // 4) residual + layernorm
    ln_kernel<<<BB, 256, 0, stream>>>(x, proj, gamma, beta, out);
}
